// TransGRUNet_1_4_89747636617848
// MI455X (gfx1250) — compile-verified, matrix operand order checked by reading, not run on hardware
//
#include <hip/hip_runtime.h>
#include <hip/hip_bf16.h>

typedef __attribute__((ext_vector_type(16))) _Float16 v16h;
typedef __attribute__((ext_vector_type(8)))  float    v8f;

#define N_NODES 32768
#define E_EDGES 262144
#define BATCHG  64
#define KCLUST  8
#define LSEQ    512
#define H       128
#define NG      (BATCHG*KCLUST)   // 512

// ---------------- helpers ----------------

__device__ __forceinline__ float wave_sum(float v) {
#pragma unroll
  for (int off = 16; off > 0; off >>= 1) v += __shfl_xor(v, off, 32);
  return v;
}

__device__ __forceinline__ unsigned flip_f(float f) {
  unsigned u = __float_as_uint(f);
  return (u & 0x80000000u) ? ~u : (u | 0x80000000u);
}
__device__ __forceinline__ float unflip_f(unsigned u) {
  u = (u & 0x80000000u) ? (u & 0x7FFFFFFFu) : ~u;
  return __uint_as_float(u);
}

// K index inside a 16x32 f16 A/B fragment (per CDNA5 ISA 7.12.2)
__device__ __forceinline__ int frag_k(int j, int half) {
  int v = j >> 1, p = j & 1;
  return ((v < 4) ? (2 * v) : (16 + 2 * (v - 4))) + 8 * half + p;
}

__device__ __forceinline__ float sigmoidf_(float x) { return 1.0f / (1.0f + expf(-x)); }

// CDNA5 async global<->LDS copies (ASYNCcnt-tracked), 16B per lane
__device__ __forceinline__ void async_load_b128(unsigned lds_addr, unsigned long long gaddr) {
  asm volatile("global_load_async_to_lds_b128 %0, %1, off" :: "v"(lds_addr), "v"(gaddr) : "memory");
}
__device__ __forceinline__ void async_store_b128(unsigned long long gaddr, unsigned lds_addr) {
  asm volatile("global_store_async_from_lds_b128 %0, %1, off" :: "v"(gaddr), "v"(lds_addr) : "memory");
}
__device__ __forceinline__ void wait_asynccnt0() {
  asm volatile("s_wait_asynccnt 0" ::: "memory");
}

// ---------------- generic 128-wide WMMA GEMM: C = A[M,128] @ W[128,128]^T + bias (+pe) ----------------
// Output either f32 (C) or f16 (C16) — exactly one non-null.

__global__ __launch_bounds__(256) void gemm128_kernel(
    const float* __restrict__ A, const float* __restrict__ W,
    const float* __restrict__ bias, float* __restrict__ C,
    _Float16* __restrict__ C16, const float* __restrict__ pe, int M)
{
  __shared__ _Float16 sA[128][128];
  __shared__ _Float16 sW[128][128];
  const int block_m = blockIdx.x * 128;
  const int tid = threadIdx.x;

  for (int i = tid; i < 128 * 128 / 4; i += 256) {
    int r = i >> 5, c = (i & 31) * 4;
    float4 fa = *(const float4*)(A + (size_t)(block_m + r) * 128 + c);
    sA[r][c+0]=(_Float16)fa.x; sA[r][c+1]=(_Float16)fa.y; sA[r][c+2]=(_Float16)fa.z; sA[r][c+3]=(_Float16)fa.w;
    float4 fw = *(const float4*)(W + (size_t)r * 128 + c);
    sW[r][c+0]=(_Float16)fw.x; sW[r][c+1]=(_Float16)fw.y; sW[r][c+2]=(_Float16)fw.z; sW[r][c+3]=(_Float16)fw.w;
  }
  __syncthreads();

  const int wave = tid >> 5, lane = tid & 31;
  const int half = lane >> 4, lidx = lane & 15;

  v8f acc[8];
#pragma unroll
  for (int nt = 0; nt < 8; ++nt)
#pragma unroll
    for (int r = 0; r < 8; ++r) acc[nt][r] = 0.0f;

#pragma unroll
  for (int kc = 0; kc < 4; ++kc) {
    v16h af;
#pragma unroll
    for (int j = 0; j < 16; ++j) af[j] = sA[wave * 16 + lidx][kc * 32 + frag_k(j, half)];
#pragma unroll
    for (int nt = 0; nt < 8; ++nt) {
      v16h bf;
#pragma unroll
      for (int j = 0; j < 16; ++j) bf[j] = sW[nt * 16 + lidx][kc * 32 + frag_k(j, half)];
      acc[nt] = __builtin_amdgcn_wmma_f32_16x16x32_f16(false, af, false, bf, (short)0, acc[nt], false, false);
    }
  }

#pragma unroll
  for (int nt = 0; nt < 8; ++nt) {
    int col = nt * 16 + lidx;
    float b = bias[col];
#pragma unroll
    for (int r = 0; r < 8; ++r) {
      int grow = block_m + wave * 16 + r + 8 * half;
      float vv = acc[nt][r] + b;
      if (pe) vv += pe[(grow & (LSEQ - 1)) * H + col];
      if (C16) C16[(size_t)grow * 128 + col] = (_Float16)vv;
      else     C[(size_t)grow * 128 + col] = vv;
    }
  }
}

// ---------------- per-layer edge attention ----------------

__global__ void init_layer_kernel(float* agg, unsigned* segmax, float* segsum) {
  size_t idx = (size_t)blockIdx.x * 256 + threadIdx.x;
  if (idx < (size_t)N_NODES * H) agg[idx] = 0.0f;
  if (idx < N_NODES) { segmax[idx] = 0u; segsum[idx] = 0.0f; }
}

__global__ void edge_alpha_kernel(const float* __restrict__ q, const float* __restrict__ k,
                                  const int* __restrict__ src, const int* __restrict__ dst,
                                  float* __restrict__ alpha, unsigned* __restrict__ segmax) {
  int e = (blockIdx.x * 256 + threadIdx.x) >> 5;
  int lane = threadIdx.x & 31;
  if (e >= E_EDGES) return;
  int s = src[e], d = dst[e];
  float4 qv = *(const float4*)(q + (size_t)d * H + lane * 4);
  float4 kv = *(const float4*)(k + (size_t)s * H + lane * 4);
  float p = wave_sum(qv.x*kv.x + qv.y*kv.y + qv.z*kv.z + qv.w*kv.w) * 0.08838834764831845f;
  if (lane == 0) { alpha[e] = p; atomicMax(segmax + d, flip_f(p)); }
}

__global__ void edge_agg_kernel(const float* __restrict__ alpha, const unsigned* __restrict__ segmax,
                                float* __restrict__ segsum, const float* __restrict__ v,
                                const int* __restrict__ src, const int* __restrict__ dst,
                                float* __restrict__ agg) {
  int e = (blockIdx.x * 256 + threadIdx.x) >> 5;
  int lane = threadIdx.x & 31;
  if (e >= E_EDGES) return;
  int s = src[e], d = dst[e];
  float ee = expf(alpha[e] - unflip_f(segmax[d]));
  if (lane == 0) atomicAdd(segsum + d, ee);
  float4 vv = *(const float4*)(v + (size_t)s * H + lane * 4);
  float* a = agg + (size_t)d * H + lane * 4;
  atomicAdd(a + 0, ee * vv.x); atomicAdd(a + 1, ee * vv.y);
  atomicAdd(a + 2, ee * vv.z); atomicAdd(a + 3, ee * vv.w);
}

__global__ void finalize_kernel(const float* __restrict__ agg, const float* __restrict__ segsum,
                                const float* __restrict__ skip, const float* __restrict__ g,
                                const float* __restrict__ b, float* __restrict__ hout) {
  int i = (blockIdx.x * 256 + threadIdx.x) >> 5;
  int lane = threadIdx.x & 31;
  if (i >= N_NODES) return;
  float s = segsum[i] + 1e-16f;
  float4 av = *(const float4*)(agg  + (size_t)i * H + lane * 4);
  float4 kv = *(const float4*)(skip + (size_t)i * H + lane * 4);
  float v0 = av.x / s + kv.x, v1 = av.y / s + kv.y, v2 = av.z / s + kv.z, v3 = av.w / s + kv.w;
  float mu = wave_sum(v0 + v1 + v2 + v3) * (1.0f / 128.0f);
  float d0 = v0 - mu, d1 = v1 - mu, d2 = v2 - mu, d3 = v3 - mu;
  float var = wave_sum(d0*d0 + d1*d1 + d2*d2 + d3*d3) * (1.0f / 128.0f);
  float inv = rsqrtf(var + 1e-5f);
  int c = lane * 4;
  float* o = hout + (size_t)i * H + c;
  o[0] = fmaxf(d0 * inv * g[c+0] + b[c+0], 0.0f);
  o[1] = fmaxf(d1 * inv * g[c+1] + b[c+1], 0.0f);
  o[2] = fmaxf(d2 * inv * g[c+2] + b[c+2], 0.0f);
  o[3] = fmaxf(d3 * inv * g[c+3] + b[c+3], 0.0f);
}

// ---------------- clustering / grouping ----------------

__global__ void init_post_kernel(int* counts, int* node_idx, float* seq_raw) {
  size_t idx = (size_t)blockIdx.x * 256 + threadIdx.x;
  if (idx < (size_t)NG * LSEQ * H) seq_raw[idx] = 0.0f;
  if (idx < (size_t)NG * LSEQ) node_idx[idx] = -1;
  if (idx <= NG) counts[idx] = 0;
}

__global__ void assign_kernel(const float* __restrict__ h, const float* __restrict__ centers,
                              const int* __restrict__ batch, int* __restrict__ gid,
                              int* __restrict__ counts) {
  int i = (blockIdx.x * 256 + threadIdx.x) >> 5;
  int lane = threadIdx.x & 31;
  if (i >= N_NODES) return;
  float4 hv = *(const float4*)(h + (size_t)i * H + lane * 4);
  float hn = sqrtf(wave_sum(hv.x*hv.x + hv.y*hv.y + hv.z*hv.z + hv.w*hv.w));
  float best = -1e30f; int bi = 0;
  for (int j = 0; j < KCLUST; ++j) {
    float4 cv = *(const float4*)(centers + (size_t)j * H + lane * 4);
    float dp = wave_sum(hv.x*cv.x + hv.y*cv.y + hv.z*cv.z + hv.w*cv.w);
    float cn = sqrtf(wave_sum(cv.x*cv.x + cv.y*cv.y + cv.z*cv.z + cv.w*cv.w));
    float sim = dp / fmaxf(hn * cn, 1e-8f);
    if (sim > best) { best = sim; bi = j; }
  }
  if (lane == 0) { int g = batch[i] * KCLUST + bi; gid[i] = g; atomicAdd(counts + g, 1); }
}

__global__ void distance_kernel(const float* __restrict__ centers, float* __restrict__ out) {
  __shared__ float pd[28];
  int t = threadIdx.x;
  if (t < 28) {
    int i = 0, j = 0, c = t;
    for (i = 0; i < 8; ++i) { int cnt = 7 - i; if (c < cnt) { j = i + 1 + c; break; } c -= cnt; }
    float s = 0.0f;
    for (int d = 0; d < H; ++d) { float df = centers[i*H+d] - centers[j*H+d]; s += df * df; }
    pd[t] = sqrtf(s);
  }
  __syncthreads();
  if (t == 0) {
    float mu = 0.0f; for (int a = 0; a < 28; ++a) mu += pd[a]; mu *= (1.0f / 28.0f);
    float v = 0.0f; for (int a = 0; a < 28; ++a) { float d = pd[a] - mu; v += d * d; }
    out[0] = -(v / 27.0f);
  }
}

// stable rank within (batch,cluster) group by (call, index) — brute force with LDS tiles
__global__ __launch_bounds__(256) void rank_kernel(const int* __restrict__ gid,
                                                   const int* __restrict__ call,
                                                   int* __restrict__ posn) {
  __shared__ int sg[256];
  __shared__ int sc[256];
  int i = blockIdx.x * 256 + threadIdx.x;
  int my_g = gid[i], my_c = call[i];
  int r = 0;
  for (int base = 0; base < N_NODES; base += 256) {
    sg[threadIdx.x] = gid[base + threadIdx.x];
    sc[threadIdx.x] = call[base + threadIdx.x];
    __syncthreads();
    for (int j = 0; j < 256; ++j) {
      int jj = base + j;
      if (sg[j] == my_g && (sc[j] < my_c || (sc[j] == my_c && jj < i))) ++r;
    }
    __syncthreads();
  }
  posn[i] = r;
}

__global__ void scatter_seq_kernel(const float* __restrict__ h, const int* __restrict__ gid,
                                   const int* __restrict__ counts, const int* __restrict__ posn,
                                   float* __restrict__ seq_raw, int* __restrict__ node_idx) {
  int i = (blockIdx.x * 256 + threadIdx.x) >> 5;
  int lane = threadIdx.x & 31;
  if (i >= N_NODES) return;
  int g = gid[i];
  if (counts[g] == 1) return;             // singleton bypasses RNN
  int p = posn[i];
  if (p >= LSEQ) return;                  // dropped by padding (JAX scatter-drop)
  size_t row = (size_t)g * LSEQ + p;
  float4 hv = *(const float4*)(h + (size_t)i * H + lane * 4);
  *(float4*)(seq_raw + row * H + lane * 4) = hv;
  if (lane == 0) node_idx[row] = i;
}

// ---------------- positional encoding & f16 weight conversion ----------------

__global__ void pe_kernel(float* pe) {
  int idx = blockIdx.x * 256 + threadIdx.x;
  if (idx >= LSEQ * 64) return;
  int t = idx >> 6, i = idx & 63;
  float div = expf((float)(2 * i) * (-9.210340371976184f / 128.0f));
  float a = (float)t * div;
  pe[t * H + 2 * i]     = sinf(a);
  pe[t * H + 2 * i + 1] = cosf(a);
}

__global__ void cvt_f16_kernel(const float* __restrict__ src, _Float16* __restrict__ dst, int n) {
  int i = blockIdx.x * 256 + threadIdx.x;
  if (i < n) dst[i] = (_Float16)src[i];
}

// ---------------- persistent WMMA GRU ----------------
// 16 sequences per workgroup; Wih/Whh fragments resident in VGPRs across the scan;
// x_t tiles prefetched with GLOBAL_LOAD_ASYNC_TO_LDS_B128 (double buffered);
// h_t tiles drained with GLOBAL_STORE_ASYNC_FROM_LDS_B128.

__global__ __launch_bounds__(256) void gru_kernel(
    const _Float16* __restrict__ sproj,       // [NG*LSEQ, H] f16
    const _Float16* __restrict__ wih,         // [384,128]
    const _Float16* __restrict__ whh,         // [384,128]
    const float* __restrict__ bih, const float* __restrict__ bhh,
    _Float16* __restrict__ hout)              // [NG*LSEQ, H] f16
{
  __shared__ _Float16 sX[2][16][128];
  __shared__ _Float16 sH[16][128];
  const int g0 = blockIdx.x * 16;
  const int tid = threadIdx.x, wave = tid >> 5, lane = tid & 31;
  const int half = lane >> 4, lidx = lane & 15;

  // staging assignment: thread -> (row, 16-byte chunk)
  const int srow = tid >> 4, schunk = tid & 15;
  const unsigned ldsX0 = (unsigned)(size_t)&sX[0][srow][schunk * 8];
  const unsigned ldsX1 = (unsigned)(size_t)&sX[1][srow][schunk * 8];
  const unsigned ldsH  = (unsigned)(size_t)&sH[srow][schunk * 8];
  const unsigned long long gx_base =
      (unsigned long long)(const void*)(sproj + (size_t)(g0 + srow) * LSEQ * H) + schunk * 16;
  const unsigned long long gh_base =
      (unsigned long long)(void*)(hout + (size_t)(g0 + srow) * LSEQ * H) + schunk * 16;

  // wave w owns N-tiles {w, w+8, w+16} -> same 16 columns of gates r, z, n
  v16h wf_ih[3][4], wf_hh[3][4];
  float b_ih[3], b_hh[3];
#pragma unroll
  for (int t = 0; t < 3; ++t) {
    int n = (wave + t * 8) * 16 + lidx;
    b_ih[t] = bih[n]; b_hh[t] = bhh[n];
#pragma unroll
    for (int kc = 0; kc < 4; ++kc)
#pragma unroll
      for (int j = 0; j < 16; ++j) {
        int k = kc * 32 + frag_k(j, half);
        wf_ih[t][kc][j] = wih[(size_t)n * 128 + k];
        wf_hh[t][kc][j] = whh[(size_t)n * 128 + k];
      }
  }

  float hreg[8];
#pragma unroll
  for (int r = 0; r < 8; ++r) hreg[r] = 0.0f;
  for (int i = tid; i < 16 * 128; i += 256) ((_Float16*)sH)[i] = (_Float16)0.0f;

  // prefetch x_0 into sX[0]
  async_load_b128(ldsX0, gx_base);

  const int col = wave * 16 + lidx;
  for (int t = 0; t < LSEQ; ++t) {
    wait_asynccnt0();     // x_t landed; h_{t-1} store drained (WAR guard on sH)
    __syncthreads();
    if (t + 1 < LSEQ) {   // prefetch x_{t+1} into the other buffer
      unsigned long long ga = gx_base + (unsigned long long)(t + 1) * (H * 2);
      async_load_b128((t & 1) ? ldsX0 : ldsX1, ga);
    }

    v8f accX[3], accH[3];
#pragma unroll
    for (int tt = 0; tt < 3; ++tt)
#pragma unroll
      for (int r = 0; r < 8; ++r) { accX[tt][r] = b_ih[tt]; accH[tt][r] = b_hh[tt]; }

    const _Float16* xrow = &sX[t & 1][lidx][0];
#pragma unroll
    for (int kc = 0; kc < 4; ++kc) {
      v16h ax, ah;
#pragma unroll
      for (int j = 0; j < 16; ++j) {
        int k = kc * 32 + frag_k(j, half);
        ax[j] = xrow[k];
        ah[j] = sH[lidx][k];
      }
#pragma unroll
      for (int tt = 0; tt < 3; ++tt) {
        accX[tt] = __builtin_amdgcn_wmma_f32_16x16x32_f16(false, ax, false, wf_ih[tt][kc], (short)0, accX[tt], false, false);
        accH[tt] = __builtin_amdgcn_wmma_f32_16x16x32_f16(false, ah, false, wf_hh[tt][kc], (short)0, accH[tt], false, false);
      }
    }
    __syncthreads();   // everyone done reading sH/sX

    // gate math is register-local: tile0=r-gate, tile1=z-gate, tile2=n-gate (PyTorch order)
#pragma unroll
    for (int r = 0; r < 8; ++r) {
      float rg = sigmoidf_(accX[0][r] + accH[0][r]);
      float zg = sigmoidf_(accX[1][r] + accH[1][r]);
      float ng = tanhf(accX[2][r] + rg * accH[2][r]);
      float hn = (1.0f - zg) * ng + zg * hreg[r];
      hreg[r] = hn;
      sH[r + 8 * half][col] = (_Float16)hn;
    }
    __syncthreads();   // sH complete -> drain h_t asynchronously from LDS
    async_store_b128(gh_base + (unsigned long long)t * (H * 2), ldsH);
  }
  wait_asynccnt0();
}

// ---------------- post-GRU LN+ReLU (f16 in place), scatter back, MLP head ----------------

__global__ void row_ln_relu_f16_kernel(_Float16* __restrict__ data, const float* __restrict__ g,
                                       const float* __restrict__ b, int M) {
  int row = (blockIdx.x * 256 + threadIdx.x) >> 5;
  int lane = threadIdx.x & 31;
  if (row >= M) return;
  _Float16* p = data + (size_t)row * H + lane * 4;
  float v0 = (float)p[0], v1 = (float)p[1], v2 = (float)p[2], v3 = (float)p[3];
  float mu = wave_sum(v0 + v1 + v2 + v3) * (1.0f / 128.0f);
  float d0 = v0 - mu, d1 = v1 - mu, d2 = v2 - mu, d3 = v3 - mu;
  float var = wave_sum(d0*d0 + d1*d1 + d2*d2 + d3*d3) * (1.0f / 128.0f);
  float inv = rsqrtf(var + 1e-5f);
  int c = lane * 4;
  p[0] = (_Float16)fmaxf(d0 * inv * g[c+0] + b[c+0], 0.0f);
  p[1] = (_Float16)fmaxf(d1 * inv * g[c+1] + b[c+1], 0.0f);
  p[2] = (_Float16)fmaxf(d2 * inv * g[c+2] + b[c+2], 0.0f);
  p[3] = (_Float16)fmaxf(d3 * inv * g[c+3] + b[c+3], 0.0f);
}

__global__ void outfeat_init_kernel(const float* __restrict__ h, const int* __restrict__ gid,
                                    const int* __restrict__ counts, float* __restrict__ feat) {
  size_t idx = (size_t)blockIdx.x * 256 + threadIdx.x;
  if (idx >= (size_t)N_NODES * H) return;
  int i = (int)(idx >> 7);
  feat[idx] = (counts[gid[i]] == 1) ? h[idx] : 0.0f;
}

__global__ void scatter_out_kernel(const _Float16* __restrict__ sout, const int* __restrict__ node_idx,
                                   float* __restrict__ feat) {
  int row = (blockIdx.x * 256 + threadIdx.x) >> 5;
  int lane = threadIdx.x & 31;
  if (row >= NG * LSEQ) return;
  int i = node_idx[row];
  if (i < 0) return;
  const _Float16* s = sout + (size_t)row * H + lane * 4;
  float* o = feat + (size_t)i * H + lane * 4;
  o[0] = (float)s[0]; o[1] = (float)s[1]; o[2] = (float)s[2]; o[3] = (float)s[3];
}

__global__ void head_kernel(const float* __restrict__ feat, const float* __restrict__ w1,
                            const float* __restrict__ b1, const float* __restrict__ w2,
                            const float* __restrict__ b2, float* __restrict__ y) {
  int i = (blockIdx.x * 256 + threadIdx.x) >> 5;
  int lane = threadIdx.x & 31;
  if (i >= N_NODES) return;
  float4 xv = *(const float4*)(feat + (size_t)i * H + lane * 4);
  float outs[8];
#pragma unroll
  for (int o = 0; o < 8; ++o) {
    float4 wv = *(const float4*)(w1 + o * H + lane * 4);
    outs[o] = wave_sum(xv.x*wv.x + xv.y*wv.y + xv.z*wv.z + xv.w*wv.w) + b1[o];
  }
  if (lane == 0) {
    float s2 = b2[0];
#pragma unroll
    for (int o = 0; o < 8; ++o) {
      float u = outs[o];
      float gl = 0.5f * u * (1.0f + erff(u * 0.7071067811865475f));
      s2 += gl * w2[o];
    }
    y[i] = sigmoidf_(s2);
  }
}

// ---------------- orchestration ----------------

extern "C" void kernel_launch(void* const* d_in, const int* in_sizes, int n_in,
                              void* d_out, int out_size, void* d_ws, size_t ws_size,
                              hipStream_t stream) {
  (void)in_sizes; (void)n_in; (void)out_size; (void)ws_size;
  const float* x       = (const float*)d_in[0];
  const int*   edge    = (const int*)d_in[1];
  const int*   call    = (const int*)d_in[2];
  const int*   batch   = (const int*)d_in[4];
  const float* wq = (const float*)d_in[5],  *bq = (const float*)d_in[6];
  const float* wk = (const float*)d_in[7],  *bk = (const float*)d_in[8];
  const float* wv = (const float*)d_in[9],  *bv = (const float*)d_in[10];
  const float* wsk= (const float*)d_in[11], *bsk= (const float*)d_in[12];
  const float* lng = (const float*)d_in[13], *lnb = (const float*)d_in[14];
  const float* centers = (const float*)d_in[15];
  const float* projw = (const float*)d_in[16], *projb = (const float*)d_in[17];
  const float* gwih = (const float*)d_in[18], *gwhh = (const float*)d_in[19];
  const float* gbih = (const float*)d_in[20], *gbhh = (const float*)d_in[21];
  const float* slng = (const float*)d_in[22], *slnb = (const float*)d_in[23];
  const float* l1w = (const float*)d_in[24], *l1b = (const float*)d_in[25];
  const float* l2w = (const float*)d_in[26], *l2b = (const float*)d_in[27];

  const int* src = edge;
  const int* dst = edge + E_EDGES;

  char* ws = (char*)d_ws;
  size_t off = 0;
  auto alloc = [&](size_t bytes) -> void* {
    void* p = (void*)(ws + off);
    off += (bytes + 255) & ~(size_t)255;
    return p;
  };
  float*    hbuf   = (float*)alloc((size_t)N_NODES * H * 4);
  float*    qbuf   = (float*)alloc((size_t)N_NODES * H * 4);
  float*    kbuf   = (float*)alloc((size_t)N_NODES * H * 4);
  float*    vbuf   = (float*)alloc((size_t)N_NODES * H * 4);
  float*    skbuf  = (float*)alloc((size_t)N_NODES * H * 4);
  float*    agg    = (float*)alloc((size_t)N_NODES * H * 4);
  float*    alpha  = (float*)alloc((size_t)E_EDGES * 4);
  unsigned* segmax = (unsigned*)alloc((size_t)N_NODES * 4);
  float*    segsum = (float*)alloc((size_t)N_NODES * 4);
  int*      gid    = (int*)alloc((size_t)N_NODES * 4);
  int*      posn   = (int*)alloc((size_t)N_NODES * 4);
  int*      counts = (int*)alloc((size_t)(NG + 1) * 4);
  int*      nidx   = (int*)alloc((size_t)NG * LSEQ * 4);
  float*    pe     = (float*)alloc((size_t)LSEQ * H * 4);
  _Float16* wih16  = (_Float16*)alloc((size_t)384 * 128 * 2);
  _Float16* whh16  = (_Float16*)alloc((size_t)384 * 128 * 2);
  float*    seqraw = (float*)alloc((size_t)NG * LSEQ * H * 4);  // f32 raw seqs; front reused as f16 GRU out
  _Float16* seqprj = (_Float16*)alloc((size_t)NG * LSEQ * H * 2); // f16 proj+pe output
  float*    feat   = (float*)alloc((size_t)N_NODES * H * 4);

  _Float16* hout16 = (_Float16*)seqraw;  // GRU output aliases raw-seq buffer (dead after proj GEMM)
  float* yout = (float*)d_out;

  pe_kernel<<<(LSEQ * 64 + 255) / 256, 256, 0, stream>>>(pe);
  cvt_f16_kernel<<<(384 * 128 + 255) / 256, 256, 0, stream>>>(gwih, wih16, 384 * 128);
  cvt_f16_kernel<<<(384 * 128 + 255) / 256, 256, 0, stream>>>(gwhh, whh16, 384 * 128);

  const int nodeWaves = N_NODES / 8;        // 4096 blocks, 1 wave/node
  const int edgeWaves = E_EDGES / 8;        // 32768 blocks
  const int seqWaves  = (NG * LSEQ) / 8;    // 32768 blocks

  for (int i = 0; i < 4; ++i) {
    const float* A = (i == 0) ? x : hbuf;
    gemm128_kernel<<<N_NODES / 128, 256, 0, stream>>>(A, wq  + i*H*H, bq  + i*H, qbuf,  nullptr, nullptr, N_NODES);
    gemm128_kernel<<<N_NODES / 128, 256, 0, stream>>>(A, wk  + i*H*H, bk  + i*H, kbuf,  nullptr, nullptr, N_NODES);
    gemm128_kernel<<<N_NODES / 128, 256, 0, stream>>>(A, wv  + i*H*H, bv  + i*H, vbuf,  nullptr, nullptr, N_NODES);
    gemm128_kernel<<<N_NODES / 128, 256, 0, stream>>>(A, wsk + i*H*H, bsk + i*H, skbuf, nullptr, nullptr, N_NODES);
    init_layer_kernel<<<(N_NODES * H + 255) / 256, 256, 0, stream>>>(agg, segmax, segsum);
    edge_alpha_kernel<<<edgeWaves, 256, 0, stream>>>(qbuf, kbuf, src, dst, alpha, segmax);
    edge_agg_kernel<<<edgeWaves, 256, 0, stream>>>(alpha, segmax, segsum, vbuf, src, dst, agg);
    finalize_kernel<<<nodeWaves, 256, 0, stream>>>(agg, segsum, skbuf, lng + i*H, lnb + i*H, hbuf);
  }

  init_post_kernel<<<(int)(((size_t)NG * LSEQ * H + 255) / 256), 256, 0, stream>>>(counts, nidx, seqraw);
  assign_kernel<<<nodeWaves, 256, 0, stream>>>(hbuf, centers, batch, gid, counts);
  rank_kernel<<<N_NODES / 256, 256, 0, stream>>>(gid, call, posn);
  scatter_seq_kernel<<<nodeWaves, 256, 0, stream>>>(hbuf, gid, counts, posn, seqraw, nidx);
  distance_kernel<<<1, 32, 0, stream>>>(centers, yout + N_NODES);

  // proj + pos-enc, f16 output
  gemm128_kernel<<<NG * LSEQ / 128, 256, 0, stream>>>(seqraw, projw, projb, nullptr, seqprj, pe, NG * LSEQ);
  // GRU scan (overwrites seqraw front half as f16 output)
  gru_kernel<<<NG / 16, 256, 0, stream>>>(seqprj, wih16, whh16, gbih, gbhh, hout16);
  row_ln_relu_f16_kernel<<<seqWaves, 256, 0, stream>>>(hout16, slng, slnb, NG * LSEQ);

  outfeat_init_kernel<<<(N_NODES * H + 255) / 256, 256, 0, stream>>>(hbuf, gid, counts, feat);
  scatter_out_kernel<<<seqWaves, 256, 0, stream>>>(hout16, nidx, feat);
  head_kernel<<<nodeWaves, 256, 0, stream>>>(feat, l1w, l1b, l2w, l2b, yout);
}
